// SpatialAttentionLayer_47974784696832
// MI455X (gfx1250) — matrix operand order repaired and checked by reading, not compiled, not run on hardware
//
#include <hip/hip_runtime.h>
#include <hip/hip_bf16.h>

// ---------------------------------------------------------------------------
// SpatialAttentionLayer fused kernel for gfx1250 (MI455X)
//   - f16 WMMA (v_wmma_f32_16x16x32_f16) for QKV projections and attention
//   - weights pre-swizzled to WMMA B-fragment order in d_ws (prologue kernel)
//   - fused residual + LayerNorm; x read once from HBM, y written once
// ---------------------------------------------------------------------------

typedef _Float16 v8h  __attribute__((ext_vector_type(8)));
typedef _Float16 v16h __attribute__((ext_vector_type(16)));
typedef float    v8f  __attribute__((ext_vector_type(8)));

#define WMMA_F16(A, B, C) \
  __builtin_amdgcn_wmma_f32_16x16x32_f16(false, (A), false, (B), (short)0, (C), false, false)

static constexpr int Tn = 8192;   // tokens
static constexpr int Nj = 24;     // joints
static constexpr int Dd = 256;    // model dim per joint
static constexpr int Hh = 8;      // heads
static constexpr int DT = 6144;   // Nj * Dd
static constexpr int TB = 16;     // tokens per workgroup (WMMA M)

// pre-swizzled f16 weights in d_ws (halfs):
//   WQs: [H][N][kt=8][nt=2][lane=32][16]   (1,572,864 halfs)
//   WKs: [H][kt=8][nt=2][lane=32][16]      (   65,536 halfs)
//   WVs: same as WKs
static constexpr int WQSZ = Hh * Nj * 8 * 2 * 512;  // 1572864
static constexpr int WKSZ = Hh * 8 * 2 * 512;       // 65536
static constexpr int WTOT = WQSZ + 2 * WKSZ;        // 1703936 halfs = 3.25 MB

// LDS layout (halfs):
//   Ys : [16][6144]          = 98304
//   Qs : [16][24][32]        = 12288
//   Ks : [16][24][32]        = 12288
//   Vs : [16][32][32] (V^T)  = 16384
//   Ps : [8 waves][32][32]   =  8192
static constexpr int LDS_YS = 0;
static constexpr int LDS_QS = LDS_YS + TB * DT;
static constexpr int LDS_KS = LDS_QS + TB * 24 * 32;
static constexpr int LDS_VS = LDS_KS + TB * 24 * 32;
static constexpr int LDS_PS = LDS_VS + TB * 32 * 32;
static constexpr int SMEM_HALFS = LDS_PS + 8 * 32 * 32;          // 147456
static constexpr size_t SMEM_BYTES = (size_t)SMEM_HALFS * 2;     // 294912 B

// ---------------------------------------------------------------------------
// Prologue: convert f32 weights -> f16, pre-swizzled into WMMA B-fragment
// order.  B (32x16 f16), k order used here (same as A): lane<16 -> N=lane,   K = s      (s=0..15)
//                               lane>=16 -> N=lane-16, K = 16+s
// stored as tile*512 + lane*16 + s  => one contiguous 32B load per lane.
// ---------------------------------------------------------------------------
__global__ void swz_weights_kernel(const float* __restrict__ Wq,
                                   const float* __restrict__ Wk,
                                   const float* __restrict__ Wv,
                                   _Float16* __restrict__ w) {
  int gid = blockIdx.x * 256 + threadIdx.x;
  if (gid >= WTOT) return;
  if (gid < WQSZ) {
    int e = gid & 511, tile = gid >> 9;
    int lanei = e >> 4, s = e & 15;
    int nt = tile & 1, kt = (tile >> 1) & 7;
    int n = (tile >> 4) % 24, h = tile / 384;
    int d = kt * 32 + (lanei >= 16 ? 16 : 0) + s;
    int f = nt * 16 + (lanei & 15);
    w[gid] = (_Float16)Wq[((h * Nj + n) * Dd + d) * 32 + f];
  } else if (gid < WQSZ + WKSZ) {
    int u = gid - WQSZ;
    int e = u & 511, tile = u >> 9;
    int lanei = e >> 4, s = e & 15;
    int nt = tile & 1, kt = (tile >> 1) & 7, h = tile >> 4;
    int d = kt * 32 + (lanei >= 16 ? 16 : 0) + s;
    int f = nt * 16 + (lanei & 15);
    w[gid] = (_Float16)Wk[(h * Dd + d) * 32 + f];
  } else {
    int u = gid - WQSZ - WKSZ;
    int e = u & 511, tile = u >> 9;
    int lanei = e >> 4, s = e & 15;
    int nt = tile & 1, kt = (tile >> 1) & 7, h = tile >> 4;
    int d = kt * 32 + (lanei >= 16 ? 16 : 0) + s;
    int f = nt * 16 + (lanei & 15);
    w[gid] = (_Float16)Wv[(h * Dd + d) * 32 + f];
  }
}

// load 16-half A fragment from row-major LDS: contiguous {g*16..+15}
__device__ inline v16h lds_afrag(const _Float16* p) {
  v8h lo = *(const v8h*)p;
  v8h hi = *(const v8h*)(p + 8);
  return __builtin_shufflevector(lo, hi, 0, 1, 2, 3, 4, 5, 6, 7,
                                 8, 9, 10, 11, 12, 13, 14, 15);
}

__device__ inline v8f zero8() {
  v8f z = {0.f, 0.f, 0.f, 0.f, 0.f, 0.f, 0.f, 0.f};
  return z;
}

// ---------------------------------------------------------------------------
// Main fused kernel.  grid = T/16 blocks, 256 threads (8 wave32).
// ---------------------------------------------------------------------------
__global__ __launch_bounds__(256, 1)
void attn_kernel(const float* __restrict__ x, const _Float16* __restrict__ w,
                 const float* __restrict__ bq, const float* __restrict__ bk,
                 const float* __restrict__ bv, const float* __restrict__ gamma,
                 const float* __restrict__ beta, float* __restrict__ out) {
  extern __shared__ _Float16 smem[];
  _Float16* Ys = smem + LDS_YS;
  _Float16* Qs = smem + LDS_QS;
  _Float16* Ks = smem + LDS_KS;
  _Float16* Vs = smem + LDS_VS;
  _Float16* Ps = smem + LDS_PS;

  const int t0   = blockIdx.x * TB;
  const int tid  = threadIdx.x;
  const int wave = tid >> 5;
  const int lane = tid & 31;
  const int l16  = lane & 15;
  const int g    = lane >> 4;

  const _Float16* wq = w;
  const _Float16* wk = w + WQSZ;
  const _Float16* wv = w + WQSZ + WKSZ;

  // zero V^T pad rows (m = 24..31) so P's zero columns never hit NaN garbage
  for (int i = tid; i < 4096; i += 256) {
    int t = i >> 8, f = (i >> 3) & 31, m = 24 + (i & 7);
    Vs[(t * 32 + f) * 32 + m] = (_Float16)0.f;
  }
  __syncthreads();

  for (int h = 0; h < Hh; ++h) {
    // ---------------- QKV projections: wave handles 3 joints ----------------
#pragma unroll 1
    for (int j = 0; j < 3; ++j) {
      const int n = wave * 3 + j;
      v8f aQ[2] = {zero8(), zero8()};
      v8f aK[2] = {zero8(), zero8()};
      v8f aV[2] = {zero8(), zero8()};
#pragma unroll 1
      for (int kt = 0; kt < 8; ++kt) {
        // A fragment: rows = 16 tokens, K = 32 of d; f32 -> f16 on the fly
        const float* xp = x + (t0 + l16) * DT + n * Dd + kt * 32 + g * 16;
        float4 a0 = *(const float4*)xp;
        float4 a1 = *(const float4*)(xp + 4);
        float4 a2 = *(const float4*)(xp + 8);
        float4 a3 = *(const float4*)(xp + 12);
        v16h A = {(_Float16)a0.x, (_Float16)a0.y, (_Float16)a0.z, (_Float16)a0.w,
                  (_Float16)a1.x, (_Float16)a1.y, (_Float16)a1.z, (_Float16)a1.w,
                  (_Float16)a2.x, (_Float16)a2.y, (_Float16)a2.z, (_Float16)a2.w,
                  (_Float16)a3.x, (_Float16)a3.y, (_Float16)a3.z, (_Float16)a3.w};
        // pre-swizzled B fragments (contiguous 32B per lane)
        const _Float16* pq = wq + (((h * Nj + n) * 8 + kt) * 2) * 512 + lane * 16;
        const _Float16* pk = wk + ((h * 8 + kt) * 2) * 512 + lane * 16;
        const _Float16* pv = wv + ((h * 8 + kt) * 2) * 512 + lane * 16;
        v16h Bq0 = *(const v16h*)pq;
        v16h Bq1 = *(const v16h*)(pq + 512);
        v16h Bk0 = *(const v16h*)pk;
        v16h Bk1 = *(const v16h*)(pk + 512);
        v16h Bv0 = *(const v16h*)pv;
        v16h Bv1 = *(const v16h*)(pv + 512);
        aQ[0] = WMMA_F16(A, Bq0, aQ[0]);
        aQ[1] = WMMA_F16(A, Bq1, aQ[1]);
        aK[0] = WMMA_F16(A, Bk0, aK[0]);
        aK[1] = WMMA_F16(A, Bk1, aK[1]);
        aV[0] = WMMA_F16(A, Bv0, aV[0]);
        aV[1] = WMMA_F16(A, Bv1, aV[1]);
      }
      // add biases, store Q[t][n][f], K[t][n][f], V^T[t][f][n] as f16
#pragma unroll
      for (int nt = 0; nt < 2; ++nt) {
        const int f = nt * 16 + l16;
        const float bqv = bq[(h * Nj + n) * 32 + f];
        const float bkv = bk[h * 32 + f];
        const float bvv = bv[h * 32 + f];
#pragma unroll
        for (int r = 0; r < 8; ++r) {
          const int tok = r + 8 * g;
          Qs[(tok * 24 + n) * 32 + f] = (_Float16)(aQ[nt][r] + bqv);
          Ks[(tok * 24 + n) * 32 + f] = (_Float16)(aK[nt][r] + bkv);
          Vs[(tok * 32 + f) * 32 + n] = (_Float16)(aV[nt][r] + bvv);
        }
      }
    }
    __syncthreads();

    // ---------------- attention: wave handles 2 tokens ----------------------
#pragma unroll 1
    for (int tt = 0; tt < 2; ++tt) {
      const int t = wave * 2 + tt;
      // Q A-fragments (rows = joints, K = f); padded rows >=24 read garbage,
      // which only affects discarded output rows.
      v16h QA[2], KB[2];
#pragma unroll
      for (int mt = 0; mt < 2; ++mt)
        QA[mt] = lds_afrag(Qs + (t * 24 + mt * 16 + l16) * 32 + g * 16);
      // K^T B-fragments: lane reads 16 consecutive f from row m
#pragma unroll
      for (int nt = 0; nt < 2; ++nt)
        KB[nt] = *(const v16h*)(Ks + (t * 24 + nt * 16 + l16) * 32 + g * 16);

      v8f S[2][2];
#pragma unroll
      for (int mt = 0; mt < 2; ++mt)
#pragma unroll
        for (int nt = 0; nt < 2; ++nt)
          S[mt][nt] = WMMA_F16(QA[mt], KB[nt], zero8());

      // softmax over m (24 valid columns); logits are tiny -> no max-subtract
      const float sc = 0.17677669529663687f;  // 1/sqrt(32)
      const bool cvalid = (l16 < 8);          // nt=1 columns m=16..23 valid
      float E0[2][8], E1[2][8];
#pragma unroll
      for (int mt = 0; mt < 2; ++mt)
#pragma unroll
        for (int r = 0; r < 8; ++r) {
          E0[mt][r] = __expf(S[mt][0][r] * sc);
          E1[mt][r] = cvalid ? __expf(S[mt][1][r] * sc) : 0.f;
        }
#pragma unroll
      for (int mt = 0; mt < 2; ++mt)
#pragma unroll
        for (int r = 0; r < 8; ++r) {
          float s = E0[mt][r] + E1[mt][r];
          s += __shfl_xor(s, 1, 16);
          s += __shfl_xor(s, 2, 16);
          s += __shfl_xor(s, 4, 16);
          s += __shfl_xor(s, 8, 16);
          const float inv = 1.f / s;
          const int nrow = mt * 16 + r + 8 * g;
          Ps[wave * 1024 + nrow * 32 + l16] = (_Float16)(E0[mt][r] * inv);
          Ps[wave * 1024 + nrow * 32 + 16 + l16] = (_Float16)(E1[mt][r] * inv);
        }

      // P @ V  (A = P rows n, K = m; B = V^T rows f read along m)
      v16h PA[2], VB[2];
#pragma unroll
      for (int mt = 0; mt < 2; ++mt)
        PA[mt] = lds_afrag(Ps + wave * 1024 + (mt * 16 + l16) * 32 + g * 16);
#pragma unroll
      for (int ft = 0; ft < 2; ++ft)
        VB[ft] = *(const v16h*)(Vs + (t * 32 + ft * 16 + l16) * 32 + g * 16);

      v8f O[2][2];
#pragma unroll
      for (int mt = 0; mt < 2; ++mt)
#pragma unroll
        for (int ft = 0; ft < 2; ++ft)
          O[mt][ft] = WMMA_F16(PA[mt], VB[ft], zero8());

      // scatter to Y tile at this head's channel slice
#pragma unroll
      for (int mt = 0; mt < 2; ++mt)
#pragma unroll
        for (int ft = 0; ft < 2; ++ft)
#pragma unroll
          for (int r = 0; r < 8; ++r) {
            const int nidx = mt * 16 + r + 8 * g;
            if (nidx < 24)
              Ys[t * DT + nidx * Dd + h * 32 + ft * 16 + l16] =
                  (_Float16)O[mt][ft][r];
          }
    }
    __syncthreads();
  }

  // ---------------- residual + LayerNorm over 6144 channels -----------------
#pragma unroll 1
  for (int tt = 0; tt < 2; ++tt) {
    const int t = wave * 2 + tt;
    const float* xrow = x + (t0 + t) * DT;
    float sum = 0.f, sumsq = 0.f;
    for (int i = lane; i < DT; i += 32) {
      const float v = xrow[i] + (float)Ys[t * DT + i];
      sum += v;
      sumsq += v * v;
    }
#pragma unroll
    for (int off = 1; off < 32; off <<= 1) {
      sum += __shfl_xor(sum, off, 32);
      sumsq += __shfl_xor(sumsq, off, 32);
    }
    const float mu = sum * (1.f / DT);
    const float var = sumsq * (1.f / DT) - mu * mu;
    const float rstd = rsqrtf(var + 1e-5f);
    float* orow = out + (t0 + t) * DT;
    for (int i = lane; i < DT; i += 32) {
      const float v = xrow[i] + (float)Ys[t * DT + i];
      orow[i] = (v - mu) * rstd * gamma[i] + beta[i];
    }
  }
}

// ---------------------------------------------------------------------------
extern "C" void kernel_launch(void* const* d_in, const int* in_sizes, int n_in,
                              void* d_out, int out_size, void* d_ws, size_t ws_size,
                              hipStream_t stream) {
  const float* x     = (const float*)d_in[0];
  const float* Wq    = (const float*)d_in[1];
  const float* bq    = (const float*)d_in[2];
  const float* Wk    = (const float*)d_in[3];
  const float* bk    = (const float*)d_in[4];
  const float* Wv    = (const float*)d_in[5];
  const float* bv    = (const float*)d_in[6];
  const float* gamma = (const float*)d_in[7];
  const float* beta  = (const float*)d_in[8];
  float* out = (float*)d_out;
  _Float16* w = (_Float16*)d_ws;

  (void)hipFuncSetAttribute((const void*)attn_kernel,
                            hipFuncAttributeMaxDynamicSharedMemorySize,
                            (int)SMEM_BYTES);

  swz_weights_kernel<<<(WTOT + 255) / 256, 256, 0, stream>>>(Wq, Wk, Wv, w);
  attn_kernel<<<Tn / TB, 256, SMEM_BYTES, stream>>>(x, w, bq, bk, bv, gamma,
                                                    beta, out);
}
